// MultiHeadSelfAttention_12463995093182
// MI455X (gfx1250) — matrix operand order repaired and checked by reading, not compiled, not run on hardware
//
#include <hip/hip_runtime.h>

// ---------------------------------------------------------------------------
// Transformer encoder layer for MI455X (gfx1250, wave32, WMMA).
// B=4, S=2048, D=768, H=8, d_k=96, D_FF=2048.
// Compute-bound (~142 GFLOP): all GEMM-like math runs through
// v_wmma_f32_16x16x32_bf16 (f32 inputs downcast to bf16, f32 accumulate).
// Attention is flash-style (no 512MB score tensor) with K/V tiles in LDS
// shared by the 8 waves of each workgroup.
// GEMM + attention use double-buffered LDS tiles: one barrier per K-step,
// global->LDS staging overlaps the WMMA issue on the other buffer.
// ---------------------------------------------------------------------------

#define B_   4
#define S_   2048
#define D_   768
#define H_   8
#define DK_  96
#define DFF_ 2048
#define M_   (B_ * S_)   // 8192 rows

typedef __attribute__((ext_vector_type(16))) __bf16 bf16x16;
typedef __attribute__((ext_vector_type(8)))  __bf16 bf16x8;
typedef __attribute__((ext_vector_type(4)))  __bf16 bf16x4;
typedef __attribute__((ext_vector_type(8)))  float  f32x8;

union Frag { bf16x16 v; bf16x8 h[2]; };

__device__ __forceinline__ bf16x16 ld_frag(const __bf16* p0, const __bf16* p1) {
  Frag u;
  u.h[0] = *(const bf16x8*)p0;
  u.h[1] = *(const bf16x8*)p1;
  return u.v;
}

// ---------------------------------------------------------------------------
// f32 -> bf16 conversion
// ---------------------------------------------------------------------------
__global__ void cvt_bf16_kernel(const float* __restrict__ in,
                                __bf16* __restrict__ out, int n) {
  int i = blockIdx.x * 256 + threadIdx.x;
  if (i < n) out[i] = (__bf16)in[i];
}

// ---------------------------------------------------------------------------
// Tiled bf16 GEMM: C[M,N] = A[M,K] @ W[K,N], f32 accumulate via WMMA.
// Block = 256 threads (8 waves), 128x128 output tile, BK=32, double-buffered.
// Wave grid 2x4: each wave owns 64x32 (4 M-tiles x 2 N-tiles of 16x16).
// Epilogue modes:
//   0: out bf16 in [B,H,S,DK] layout, value = (acc + bias) * scale   (QKV)
//   1: out f32,  value = acc + bias + resid[row*N+col]               (Wo, FF2)
//   2: out bf16, value = relu(acc + bias)                            (FF1)
// ---------------------------------------------------------------------------
#define BM 128
#define BN 128
#define BK 32

__global__ __launch_bounds__(256)
void gemm_bf16_kernel(const __bf16* __restrict__ A, const __bf16* __restrict__ W,
                      const float* __restrict__ bias, const float* __restrict__ resid,
                      void* __restrict__ out, int M, int N, int K,
                      int mode, float scale)
{
  __shared__ __bf16 As[2 * BM * BK];    // [buf][m][k]   16 KB
  __shared__ __bf16 Bts[2 * BN * BK];   // [buf][n][k]   16 KB (W transposed)

  const int tid  = threadIdx.x;
  const int m0   = blockIdx.y * BM;
  const int n0   = blockIdx.x * BN;
  const int lane = tid & 31;
  const int ln   = lane & 15;
  const int hi   = lane >> 4;        // half-wave select
  const int wv   = tid >> 5;
  const int wr   = wv >> 2;          // 0..1 (wave row)
  const int wc   = wv & 3;           // 0..3 (wave col)

  const f32x8 zero = {0.f, 0.f, 0.f, 0.f, 0.f, 0.f, 0.f, 0.f};
  f32x8 acc[4][2];
#pragma unroll
  for (int mt = 0; mt < 4; ++mt)
#pragma unroll
    for (int nt = 0; nt < 2; ++nt) acc[mt][nt] = zero;

  const int arow = tid >> 1;         // 0..127
  const int acb  = (tid & 1) * 16;
  const int brow = tid >> 3;         // 0..31
  const int bcb  = (tid & 7) * 16;

  // stage one BKxBM / BKxBN tile pair into LDS buffer `buf`
  auto stage = [&](int buf, int kk) {
    __bf16* as = As  + buf * (BM * BK);
    __bf16* bs = Bts + buf * (BN * BK);
    {
      const __bf16* gp = A + (size_t)(m0 + arow) * K + kk + acb;
      bf16x8 a0 = *(const bf16x8*)gp;
      bf16x8 a1 = *(const bf16x8*)(gp + 8);
      *(bf16x8*)&as[arow * BK + acb]     = a0;
      *(bf16x8*)&as[arow * BK + acb + 8] = a1;
    }
    {
      const __bf16* gp = W + (size_t)(kk + brow) * N + n0 + bcb;
      Frag u;
      u.h[0] = *(const bf16x8*)gp;
      u.h[1] = *(const bf16x8*)(gp + 8);
#pragma unroll
      for (int j = 0; j < 16; ++j)
        bs[(bcb + j) * BK + brow] = u.v[j];
    }
  };

  stage(0, 0);
  int cur = 0;

  for (int kk = 0; kk < K; kk += BK) {
    __syncthreads();                       // staged data for `cur` visible
    if (kk + BK < K) {
      stage(cur ^ 1, kk + BK);             // overlaps with WMMAs below
      if (kk + 2 * BK < K) {
        __builtin_prefetch(A + (size_t)(m0 + arow) * K + kk + 2 * BK + acb, 0, 3);
        __builtin_prefetch(W + (size_t)(kk + 2 * BK + brow) * N + n0 + bcb, 0, 3);
      }
    }

    const __bf16* as = As  + cur * (BM * BK);
    const __bf16* bs = Bts + cur * (BN * BK);
    bf16x16 af[4], bfr[2];
#pragma unroll
    for (int mt = 0; mt < 4; ++mt) {
      const int r  = wr * 64 + mt * 16 + ln;
      const int d0 = hi * 8;
      af[mt] = ld_frag(&as[r * BK + d0], &as[r * BK + d0 + 16]);
    }
#pragma unroll
    for (int nt = 0; nt < 2; ++nt) {
      const int n  = wc * 32 + nt * 16 + ln;
      const int k0 = hi * 8;
      bfr[nt] = ld_frag(&bs[n * BK + k0], &bs[n * BK + k0 + 16]);
    }
#pragma unroll
    for (int mt = 0; mt < 4; ++mt)
#pragma unroll
      for (int nt = 0; nt < 2; ++nt)
        acc[mt][nt] = __builtin_amdgcn_wmma_f32_16x16x32_bf16(
            false, af[mt], false, bfr[nt], (short)0, acc[mt][nt], false, false);
    cur ^= 1;
  }

  // ---- epilogue ----
#pragma unroll
  for (int mt = 0; mt < 4; ++mt) {
#pragma unroll
    for (int nt = 0; nt < 2; ++nt) {
      const int colg = n0 + wc * 32 + nt * 16 + ln;
      const float bv = bias[colg];
#pragma unroll
      for (int r = 0; r < 8; ++r) {
        const int rowg = m0 + wr * 64 + mt * 16 + r + hi * 8;
        float v = acc[mt][nt][r] + bv;
        if (mode == 0) {
          v *= scale;
          const int b  = rowg >> 11;          // rowg / S_
          const int s  = rowg & (S_ - 1);
          const int hd = colg / DK_;
          const int dk = colg - hd * DK_;
          ((__bf16*)out)[(((size_t)(b * H_ + hd)) * S_ + s) * DK_ + dk] = (__bf16)v;
        } else if (mode == 1) {
          const size_t idx = (size_t)rowg * N + colg;
          ((float*)out)[idx] = v + resid[idx];
        } else {
          const size_t idx = (size_t)rowg * N + colg;
          ((__bf16*)out)[idx] = (__bf16)(v > 0.f ? v : 0.f);
        }
      }
    }
  }
}

// ---------------------------------------------------------------------------
// Flash attention. grid = (S/128, B*H); 256 threads = 8 waves.
// Wave w owns 16 query rows; all waves share K/V tiles (32 keys) in LDS,
// double-buffered (one barrier per key-step). Q pre-scaled by 1/sqrt(d_k).
// Per 32-key step: 6 WMMAs for scores (d_k=96), 6 WMMAs for P@V (d=96).
// Output written as bf16 in [B, S, H*DK] layout (ready for the Wo GEMM).
// ---------------------------------------------------------------------------
#define KVT (32 * DK_)   // elements per K or V tile (3072)

__global__ __launch_bounds__(256)
void flash_attn_kernel(const __bf16* __restrict__ Q, const __bf16* __restrict__ Kg,
                       const __bf16* __restrict__ Vg, __bf16* __restrict__ O)
{
  __shared__ __bf16 Ks[2 * KVT];     // [buf][key][d]   12 KB
  __shared__ __bf16 Vts[2 * KVT];    // [buf][d][key]   12 KB (transposed)
  __shared__ __bf16 Ps[8 * 16 * 32]; // per-wave P       8 KB

  const int tid  = threadIdx.x;
  const int lane = tid & 31;
  const int ln   = lane & 15;
  const int hi   = lane >> 4;
  const int wv   = tid >> 5;
  const int bh   = blockIdx.y;                   // b*H + h
  const int q0   = blockIdx.x * 128 + wv * 16;   // wave's query base
  const size_t headbase = (size_t)bh * S_ * DK_;

  // Q fragments (A layout, 3 chunks of K-dim=32 covering d_k=96)
  bf16x16 qf[3];
  {
    const __bf16* qp = Q + headbase + (size_t)(q0 + ln) * DK_;
    const int d0 = hi * 8;
#pragma unroll
    for (int f = 0; f < 3; ++f)
      qf[f] = ld_frag(qp + f * 32 + d0, qp + f * 32 + d0 + 16);
  }

  const f32x8 zero = {0.f, 0.f, 0.f, 0.f, 0.f, 0.f, 0.f, 0.f};
  f32x8 oacc[6];
#pragma unroll
  for (int t = 0; t < 6; ++t) oacc[t] = zero;
  float mrow[8], lrow[8];
#pragma unroll
  for (int r = 0; r < 8; ++r) { mrow[r] = -3.0e38f; lrow[r] = 0.f; }

  __bf16* Pw = Ps + wv * (16 * 32);

  // cooperatively stage K (row-major) and V (transposed) for key tile `kt`
  auto stage = [&](int buf, int kt) {
    __bf16* ks = Ks  + buf * KVT;
    __bf16* vs = Vts + buf * KVT;
    const __bf16* kbase = Kg + headbase + (size_t)kt * DK_;
    const __bf16* vbase = Vg + headbase + (size_t)kt * DK_;
#pragma unroll
    for (int j = 0; j < 3; ++j) {
      const int c = tid + j * 256;                 // 768 x bf16x4
      ((bf16x4*)ks)[c] = ((const bf16x4*)kbase)[c];
    }
#pragma unroll
    for (int j = 0; j < 3; ++j) {
      const int c = tid * 3 + j;
      bf16x4 v = ((const bf16x4*)vbase)[c];
      const int e = c * 4;
      const int k = e / DK_;
      const int d = e - k * DK_;                   // DK_ % 4 == 0: no row cross
#pragma unroll
      for (int x = 0; x < 4; ++x) vs[(d + x) * 32 + k] = v[x];
    }
  };

  stage(0, 0);
  int cur = 0;

  for (int kt = 0; kt < S_; kt += 32) {
    __syncthreads();                         // tile `cur` visible to all waves
    if (kt + 32 < S_) stage(cur ^ 1, kt + 32);   // overlaps compute below

    const __bf16* ks = Ks  + cur * KVT;
    const __bf16* vs = Vts + cur * KVT;

    // ---- scores: two 16x16 tiles (keys kt..+15, kt+16..+31) ----
    f32x8 st[2];
#pragma unroll
    for (int t = 0; t < 2; ++t) {
      st[t] = zero;
#pragma unroll
      for (int f = 0; f < 3; ++f) {
        const __bf16* kp = &ks[(t * 16 + ln) * DK_ + f * 32 + hi * 8];
        bf16x16 kf = ld_frag(kp, kp + 16);
        st[t] = __builtin_amdgcn_wmma_f32_16x16x32_bf16(
            false, qf[f], false, kf, (short)0, st[t], false, false);
      }
    }

    // ---- online softmax (row reductions across 16 lanes of each half) ----
#pragma unroll
    for (int r = 0; r < 8; ++r) {
      float s0 = st[0][r], s1 = st[1][r];
      float mx = fmaxf(s0, s1);
#pragma unroll
      for (int m = 8; m >= 1; m >>= 1) mx = fmaxf(mx, __shfl_xor(mx, m, 32));
      const float mnew = fmaxf(mrow[r], mx);
      const float p0 = __expf(s0 - mnew);
      const float p1 = __expf(s1 - mnew);
      float ps = p0 + p1;
#pragma unroll
      for (int m = 8; m >= 1; m >>= 1) ps += __shfl_xor(ps, m, 32);
      const float alpha = __expf(mrow[r] - mnew);
      lrow[r] = lrow[r] * alpha + ps;
      mrow[r] = mnew;
#pragma unroll
      for (int t = 0; t < 6; ++t) oacc[t][r] *= alpha;
      const int prow = r + hi * 8;
      Pw[prow * 32 + ln]      = (__bf16)p0;   // C layout -> LDS
      Pw[prow * 32 + 16 + ln] = (__bf16)p1;
    }

    // ---- P @ V (read P back in A layout, V as transposed B fragments) ----
    {
      const __bf16* pp = &Pw[ln * 32 + hi * 8];
      bf16x16 pf = ld_frag(pp, pp + 16);
#pragma unroll
      for (int t = 0; t < 6; ++t) {
        const __bf16* vp = &vs[(t * 16 + ln) * 32 + hi * 8];
        bf16x16 vf = ld_frag(vp, vp + 16);
        oacc[t] = __builtin_amdgcn_wmma_f32_16x16x32_bf16(
            false, pf, false, vf, (short)0, oacc[t], false, false);
      }
    }
    cur ^= 1;
  }

  // ---- finalize: O /= l, write [B, S, H*DK] bf16 ----
  const int b = bh >> 3, hd = bh & 7;
#pragma unroll
  for (int t = 0; t < 6; ++t) {
#pragma unroll
    for (int r = 0; r < 8; ++r) {
      const float v = oacc[t][r] / lrow[r];
      const int srow = q0 + r + hi * 8;
      O[((size_t)b * S_ + srow) * D_ + hd * DK_ + t * 16 + ln] = (__bf16)v;
    }
  }
}

// ---------------------------------------------------------------------------
// Row LayerNorm: one block (256 threads) per row of 768.
// Optionally also writes a bf16 copy (input to the next GEMM).
// ---------------------------------------------------------------------------
__global__ __launch_bounds__(256)
void layernorm_kernel(const float* __restrict__ y, const float* __restrict__ g,
                      const float* __restrict__ be, float* __restrict__ outf,
                      __bf16* __restrict__ outb)
{
  __shared__ float red[8];
  const int row = blockIdx.x, tid = threadIdx.x;
  const float* yr = y + (size_t)row * D_;
  float v[3];
  float s = 0.f;
#pragma unroll
  for (int j = 0; j < 3; ++j) { v[j] = yr[tid + j * 256]; s += v[j]; }
#pragma unroll
  for (int m = 16; m >= 1; m >>= 1) s += __shfl_xor(s, m, 32);
  if ((tid & 31) == 0) red[tid >> 5] = s;
  __syncthreads();
  float tot = 0.f;
#pragma unroll
  for (int j = 0; j < 8; ++j) tot += red[j];
  const float mean = tot * (1.0f / D_);
  float d2 = 0.f;
#pragma unroll
  for (int j = 0; j < 3; ++j) { const float d = v[j] - mean; d2 += d * d; }
#pragma unroll
  for (int m = 16; m >= 1; m >>= 1) d2 += __shfl_xor(d2, m, 32);
  __syncthreads();
  if ((tid & 31) == 0) red[tid >> 5] = d2;
  __syncthreads();
  float vtot = 0.f;
#pragma unroll
  for (int j = 0; j < 8; ++j) vtot += red[j];
  const float rinv = rsqrtf(vtot * (1.0f / D_) + 1e-5f);
#pragma unroll
  for (int j = 0; j < 3; ++j) {
    const int c = tid + j * 256;
    const float o = (v[j] - mean) * rinv * g[c] + be[c];
    outf[(size_t)row * D_ + c] = o;
    if (outb) outb[(size_t)row * D_ + c] = (__bf16)o;
  }
}

// ---------------------------------------------------------------------------
// Launch sequence (workspace use ~163 MB)
// ---------------------------------------------------------------------------
extern "C" void kernel_launch(void* const* d_in, const int* in_sizes, int n_in,
                              void* d_out, int out_size, void* d_ws, size_t ws_size,
                              hipStream_t stream) {
  (void)in_sizes; (void)n_in; (void)out_size; (void)ws_size;
  const float* x   = (const float*)d_in[0];
  const float* Wq  = (const float*)d_in[1];
  const float* bq  = (const float*)d_in[2];
  const float* Wk  = (const float*)d_in[3];
  const float* bk  = (const float*)d_in[4];
  const float* Wv  = (const float*)d_in[5];
  const float* bv  = (const float*)d_in[6];
  const float* Wo  = (const float*)d_in[7];
  const float* bo  = (const float*)d_in[8];
  const float* g1  = (const float*)d_in[9];
  const float* be1 = (const float*)d_in[10];
  const float* W1  = (const float*)d_in[11];
  const float* b1  = (const float*)d_in[12];
  const float* W2  = (const float*)d_in[13];
  const float* b2  = (const float*)d_in[14];
  const float* g2  = (const float*)d_in[15];
  const float* be2 = (const float*)d_in[16];
  float* out = (float*)d_out;

  char* p = (char*)d_ws;
  auto carve = [&](size_t bytes) -> char* {
    char* q = p;
    p += (bytes + 255) & ~size_t(255);
    return q;
  };
  __bf16* xb  = (__bf16*)carve((size_t)M_ * D_ * 2);
  __bf16* wqb = (__bf16*)carve((size_t)D_ * D_ * 2);
  __bf16* wkb = (__bf16*)carve((size_t)D_ * D_ * 2);
  __bf16* wvb = (__bf16*)carve((size_t)D_ * D_ * 2);
  __bf16* wob = (__bf16*)carve((size_t)D_ * D_ * 2);
  __bf16* w1b = (__bf16*)carve((size_t)D_ * DFF_ * 2);
  __bf16* w2b = (__bf16*)carve((size_t)DFF_ * D_ * 2);
  __bf16* qb  = (__bf16*)carve((size_t)M_ * D_ * 2);   // [B,H,S,DK]
  __bf16* kb  = (__bf16*)carve((size_t)M_ * D_ * 2);
  __bf16* vb  = (__bf16*)carve((size_t)M_ * D_ * 2);
  __bf16* ao  = (__bf16*)carve((size_t)M_ * D_ * 2);   // attention out [B,S,D]
  float*  y1  = (float*)carve((size_t)M_ * D_ * 4);    // reused for y2
  float*  x1f = (float*)carve((size_t)M_ * D_ * 4);
  __bf16* x1b = (__bf16*)carve((size_t)M_ * D_ * 2);
  __bf16* hb  = (__bf16*)carve((size_t)M_ * DFF_ * 2);

  const dim3 blk(256);
  auto cvt = [&](const float* src, __bf16* dst, int n) {
    cvt_bf16_kernel<<<dim3((n + 255) / 256), blk, 0, stream>>>(src, dst, n);
  };
  cvt(x,  xb,  M_ * D_);
  cvt(Wq, wqb, D_ * D_);
  cvt(Wk, wkb, D_ * D_);
  cvt(Wv, wvb, D_ * D_);
  cvt(Wo, wob, D_ * D_);
  cvt(W1, w1b, D_ * DFF_);
  cvt(W2, w2b, DFF_ * D_);

  const dim3 gD(D_ / BN, M_ / BM);     // (6, 64)
  const dim3 gF(DFF_ / BN, M_ / BM);   // (16, 64)
  const float qscale = 0.1020620726159658f;  // 1/sqrt(96), folded into Q

  // QKV projections (bias fused, Q pre-scaled, [B,H,S,DK] transpose fused)
  gemm_bf16_kernel<<<gD, blk, 0, stream>>>(xb, wqb, bq, nullptr, qb, M_, D_, D_, 0, qscale);
  gemm_bf16_kernel<<<gD, blk, 0, stream>>>(xb, wkb, bk, nullptr, kb, M_, D_, D_, 0, 1.0f);
  gemm_bf16_kernel<<<gD, blk, 0, stream>>>(xb, wvb, bv, nullptr, vb, M_, D_, D_, 0, 1.0f);

  // Flash attention
  flash_attn_kernel<<<dim3(S_ / 128, B_ * H_), blk, 0, stream>>>(qb, kb, vb, ao);

  // Output projection + residual, then LayerNorm1 (f32 + bf16 copies)
  gemm_bf16_kernel<<<gD, blk, 0, stream>>>(ao, wob, bo, x, y1, M_, D_, D_, 1, 1.0f);
  layernorm_kernel<<<dim3(M_), blk, 0, stream>>>(y1, g1, be1, x1f, x1b);

  // FFN: relu(x1 @ W1 + b1) @ W2 + b2 + x1, then LayerNorm2 -> out
  gemm_bf16_kernel<<<gF, blk, 0, stream>>>(x1b, w1b, b1, nullptr, hb, M_, DFF_, D_, 2, 1.0f);
  gemm_bf16_kernel<<<gD, blk, 0, stream>>>(hb, w2b, b2, x1f, y1, M_, D_, DFF_, 1, 1.0f);
  layernorm_kernel<<<dim3(M_), blk, 0, stream>>>(y1, g2, be2, out, nullptr);
}
